// SOM_32126355374895
// MI455X (gfx1250) — matrix operand order repaired and checked by reading, not compiled, not run on hardware
//
#include <hip/hip_runtime.h>
#include <stdint.h>

// ---------------- types for CDNA5 WMMA ----------------
typedef __attribute__((ext_vector_type(16))) __bf16 v16bfx;
typedef __attribute__((ext_vector_type(8)))  float  v8fx;

union FragAB { v16bfx v; uint4 q[2]; };

// Problem constants (match reference)
#define BATCH_N   16384
#define IN_DIM    1024
#define KNEUR     1024      // M*N = 32*32
#define INV_B     (1.0f / 16384.0f)

// =====================================================================
// init: zero S accumulator, counts, loss
// =====================================================================
__global__ void init_kernel(float* S, float* counts, float* loss) {
    int idx = blockIdx.x * 256 + threadIdx.x;
    if (idx < 1048576)            S[idx] = 0.0f;
    else if (idx < 1048576+1024)  counts[idx - 1048576] = 0.0f;
    else if (idx == 1048576+1024) loss[0] = 0.0f;
}

// =====================================================================
// prep_x: convert x rows to bf16 + compute ||x_b||^2
// =====================================================================
__global__ __launch_bounds__(256) void prep_x_kernel(const float* __restrict__ x,
                                                     __bf16* __restrict__ xbf,
                                                     float* __restrict__ x2) {
    __shared__ float red[256];
    const int b = blockIdx.x;
    const float* xr = x + (size_t)b * IN_DIM;
    __bf16* o = xbf + (size_t)b * IN_DIM;
    float acc = 0.0f;
    for (int i = threadIdx.x; i < IN_DIM; i += 256) {
        float v = xr[i];
        o[i] = (__bf16)v;
        acc += v * v;
    }
    red[threadIdx.x] = acc; __syncthreads();
    for (int s = 128; s > 0; s >>= 1) {
        if (threadIdx.x < s) red[threadIdx.x] += red[threadIdx.x + s];
        __syncthreads();
    }
    if (threadIdx.x == 0) x2[b] = red[0];
}

// =====================================================================
// prep_w: Wt[n][i] = bf16(W[i][n])  (transpose for B-fragment layout),
//         w2[n] = sum_i W[i][n]^2
// =====================================================================
__global__ __launch_bounds__(256) void prep_w_kernel(const float* __restrict__ W,
                                                     __bf16* __restrict__ Wt,
                                                     float* __restrict__ w2) {
    __shared__ float red[256];
    const int n = blockIdx.x;
    float acc = 0.0f;
    for (int i = threadIdx.x; i < IN_DIM; i += 256) {
        float v = W[(size_t)i * KNEUR + n];
        Wt[(size_t)n * IN_DIM + i] = (__bf16)v;
        acc += v * v;
    }
    red[threadIdx.x] = acc; __syncthreads();
    for (int s = 128; s > 0; s >>= 1) {
        if (threadIdx.x < s) red[threadIdx.x] += red[threadIdx.x + s];
        __syncthreads();
    }
    if (threadIdx.x == 0) w2[n] = red[0];
}

// =====================================================================
// prep_T: neighborhood table T[j][k] = lr*exp(-griddist2(j,k)/sigma^2)
//         also Tb[k][j] = bf16(T[j][k])  (B-fragment layout for GEMM2)
// =====================================================================
__global__ __launch_bounds__(256) void prep_T_kernel(const int* __restrict__ loc,
                                                     const int* __restrict__ itp,
                                                     float* __restrict__ T,
                                                     __bf16* __restrict__ Tb) {
    int idx = blockIdx.x * 256 + threadIdx.x;
    if (idx >= KNEUR * KNEUR) return;
    const float icorr = 1.0f - (float)(*itp) * 0.01f;   // 1 - iter/100
    const float lr = 0.3f * icorr;
    const float sg = 16.0f * icorr;
    const float inv_s2 = 1.0f / (sg * sg);
    int j = idx >> 10, k = idx & 1023;
    int dx = loc[2*j]   - loc[2*k];
    int dy = loc[2*j+1] - loc[2*k+1];
    float d2 = (float)(dx*dx + dy*dy);
    float t = lr * __expf(-d2 * inv_s2);
    T[idx] = t;
    Tb[(size_t)k * KNEUR + j] = (__bf16)t;
}

// =====================================================================
// Shared 128x128-tile bf16 WMMA GEMM core.
//   A (global): [row][k] bf16, 1024 k per row (512 dwords)
//   B (global): [n][k]  bf16, 1024 k per row (512 dwords)
// 8 waves: wave = (wrow in 0..3) x (wcol in 0..1); wave owns 32 rows x 64 cols
// K staged 32-wide, double buffered in LDS via async global->LDS copies
// (GLOBAL_LOAD_ASYNC_TO_LDS_B128, ASYNCcnt). The k-loop is kept rolled
// (#pragma unroll 1) with a runtime buffer offset so accumulators keep one
// register home (no C-operand shuffling / hazard nops around the WMMAs).
// MODE 0: distance epilogue -> partial (min,argmin) per row per column-block
// MODE 1: weight-update epilogue -> out = W + (acc - W*csum)/B
// =====================================================================
template <int MODE>
__global__ __launch_bounds__(256) void gemm_tile_kernel(
        const uint32_t* __restrict__ Aglob,
        const uint32_t* __restrict__ Bglob,
        const float* __restrict__ e0,     // MODE0: x2   | MODE1: W (f32)
        const float* __restrict__ e1,     // MODE0: w2   | MODE1: csum
        float* __restrict__ o0,           // MODE0: partval | MODE1: newW
        int*   __restrict__ o1)           // MODE0: partidx | MODE1: unused
{
    __shared__ uint32_t As[2 * 2048];   // 2 buffers x (128 rows x 16 dwords)
    __shared__ uint32_t Bs[2 * 2048];   // 2 buffers x (128 n    x 16 dwords)

    const int tid  = threadIdx.x;
    const int L    = tid & 31;
    const int wid  = tid >> 5;
    const int wrow = wid & 3;          // 4 row groups of 32
    const int wcol = wid >> 2;         // 2 col groups of 64
    const int row0 = blockIdx.y << 7;
    const int n0   = blockIdx.x << 7;

    v8fx acc[2][4] = {};

    // Async copy one 128x32(bf16) K-slab of A and B into LDS at dword
    // offset bufo. Each thread moves 2x16B of A and 2x16B of B; the
    // instruction offset: applies to both global and LDS addresses.
    auto stage = [&](int kt, int bufo) {
        const int mrow = tid >> 1;
        const int hh   = (tid & 1) << 3;            // dword half: 0 or 8
        uint64_t ga = (uint64_t)(uintptr_t)(Aglob + ((size_t)(row0 + mrow) << 9) + (kt << 4) + hh);
        uint32_t la = (uint32_t)(uintptr_t)(&As[bufo + (mrow << 4) + hh]);
        asm volatile("global_load_async_to_lds_b128 %0, %1, off\n\t"
                     "global_load_async_to_lds_b128 %0, %1, off offset:16"
                     :: "v"(la), "v"(ga) : "memory");
        uint64_t gb = (uint64_t)(uintptr_t)(Bglob + ((size_t)(n0 + mrow) << 9) + (kt << 4) + hh);
        uint32_t lb = (uint32_t)(uintptr_t)(&Bs[bufo + (mrow << 4) + hh]);
        asm volatile("global_load_async_to_lds_b128 %0, %1, off\n\t"
                     "global_load_async_to_lds_b128 %0, %1, off offset:16"
                     :: "v"(lb), "v"(gb) : "memory");
    };

    auto wait_async = [&]() {
        asm volatile("s_wait_asynccnt 0x0" ::: "memory");
    };

    auto compute = [&](int bufo) {
        FragAB af[2], bfg[4];
        // A frag: lane L -> M = L%16 ; VGPR r -> K = (r>=4?16:0)+(L/16)*8+(r%4)*2
        #pragma unroll
        for (int a = 0; a < 2; ++a) {
            int m = wrow * 32 + a * 16 + (L & 15);
            const uint32_t* p = &As[bufo + (m << 4) + ((L >> 4) << 2)];
            af[a].q[0] = *(const uint4*)p;
            af[a].q[1] = *(const uint4*)(p + 8);
        }
        // B frag: lane L -> N = L%16 ; VGPR r -> K = (r>=4?16:0)+(L/16)*8+(r%4)*2
        #pragma unroll
        for (int c = 0; c < 4; ++c) {
            int n = wcol * 64 + c * 16 + (L & 15);
            const uint32_t* p = &Bs[bufo + (n << 4) + ((L >> 4) << 2)];
            bfg[c].q[0] = *(const uint4*)p;
            bfg[c].q[1] = *(const uint4*)(p + 8);
        }
        #pragma unroll
        for (int a = 0; a < 2; ++a)
            #pragma unroll
            for (int c = 0; c < 4; ++c)
                acc[a][c] = __builtin_amdgcn_wmma_f32_16x16x32_bf16(
                    false, af[a].v, false, bfg[c].v, (short)0, acc[a][c], false, false);
    };

    stage(0, 0);
    wait_async();
    __syncthreads();
    int cur = 0;
    #pragma unroll 1
    for (int kt = 0; kt < 31; ++kt) {
        const int nxt = cur ^ 2048;
        stage(kt + 1, nxt);       // async copy overlaps WMMA below
        compute(cur);
        wait_async();
        __syncthreads();
        cur = nxt;
    }
    compute(cur);                 // peeled last K-step

    // C/D layout: VGPR r, lane L -> M = r + 8*(L/16), N = L%16
    if (MODE == 0) {
        #pragma unroll
        for (int a = 0; a < 2; ++a) {
            #pragma unroll
            for (int r = 0; r < 8; ++r) {
                const int mrow = row0 + wrow * 32 + a * 16 + r + ((L >> 4) << 3);
                const float xm = e0[mrow];
                float best = 3.0e38f; int bidx = 0;
                #pragma unroll
                for (int c = 0; c < 4; ++c) {
                    int n = n0 + wcol * 64 + c * 16 + (L & 15);
                    float d2 = xm + e1[n] - 2.0f * acc[a][c][r];
                    float d  = sqrtf(fmaxf(d2, 0.0f));
                    if (d < best) { best = d; bidx = n; }
                }
                #pragma unroll
                for (int off = 1; off < 16; off <<= 1) {
                    float ov = __shfl_xor(best, off, 32);
                    int   oi = __shfl_xor(bidx, off, 32);
                    if (ov < best || (ov == best && oi < bidx)) { best = ov; bidx = oi; }
                }
                if ((L & 15) == 0) {
                    int cb = blockIdx.x * 2 + wcol;          // 16 column blocks
                    o0[(size_t)cb * BATCH_N + mrow] = best;
                    o1[(size_t)cb * BATCH_N + mrow] = bidx;
                }
            }
        }
    } else {
        #pragma unroll
        for (int a = 0; a < 2; ++a) {
            #pragma unroll
            for (int c = 0; c < 4; ++c) {
                const int k  = n0 + wcol * 64 + c * 16 + (L & 15);
                const float cs = e1[k];
                #pragma unroll
                for (int r = 0; r < 8; ++r) {
                    const int i = row0 + wrow * 32 + a * 16 + r + ((L >> 4) << 3);
                    const float wv = e0[(size_t)i * KNEUR + k];
                    o0[(size_t)i * KNEUR + k] = wv + (acc[a][c][r] - wv * cs) * INV_B;
                }
            }
        }
    }
}

// =====================================================================
// combine: reduce 16 column-block partials -> BMU index; sum loss
// =====================================================================
__global__ __launch_bounds__(256) void combine_kernel(const float* __restrict__ pv,
                                                      const int* __restrict__ pi,
                                                      int* __restrict__ bmu,
                                                      float* __restrict__ loss) {
    __shared__ float red[256];
    const int b = blockIdx.x * 256 + threadIdx.x;
    float best = 3.4e38f; int bi = 0;
    for (int cb = 0; cb < 16; ++cb) {
        float v = pv[(size_t)cb * BATCH_N + b];
        int  ii = pi[(size_t)cb * BATCH_N + b];
        if (v < best || (v == best && ii < bi)) { best = v; bi = ii; }
    }
    bmu[b] = bi;
    red[threadIdx.x] = best * INV_B; __syncthreads();
    for (int s = 128; s > 0; s >>= 1) {
        if (threadIdx.x < s) red[threadIdx.x] += red[threadIdx.x + s];
        __syncthreads();
    }
    if (threadIdx.x == 0) atomicAdd(loss, red[0]);
}

// =====================================================================
// bucket: S[bmu[b]][:] += x[b][:]  ; counts[bmu[b]] += 1
// =====================================================================
__global__ __launch_bounds__(256) void bucket_kernel(const float* __restrict__ x,
                                                     const int* __restrict__ bmu,
                                                     float* __restrict__ S,
                                                     float* __restrict__ counts) {
    const int b = blockIdx.x;
    const int j = bmu[b];
    const float* xr = x + (size_t)b * IN_DIM;
    float* Sr = S + (size_t)j * IN_DIM;
    for (int i = threadIdx.x; i < IN_DIM; i += 256)
        atomicAdd(&Sr[i], xr[i]);
    if (threadIdx.x == 0) atomicAdd(&counts[j], 1.0f);
}

// =====================================================================
// convS: St[i][j] = bf16(S[j][i])  (A-fragment layout for GEMM2)
// =====================================================================
__global__ __launch_bounds__(256) void convS_kernel(const float* __restrict__ S,
                                                    __bf16* __restrict__ St) {
    int idx = blockIdx.x * 256 + threadIdx.x;
    if (idx >= KNEUR * IN_DIM) return;
    int i = idx >> 10, j = idx & 1023;
    St[idx] = (__bf16)S[(size_t)j * IN_DIM + i];
}

// =====================================================================
// csum[k] = sum_j counts[j] * T[j][k]
// =====================================================================
__global__ __launch_bounds__(256) void csum_kernel(const float* __restrict__ T,
                                                   const float* __restrict__ counts,
                                                   float* __restrict__ csum) {
    __shared__ float red[256];
    const int k = blockIdx.x;
    float acc = 0.0f;
    for (int j = threadIdx.x; j < KNEUR; j += 256)
        acc += counts[j] * T[(size_t)j * KNEUR + k];
    red[threadIdx.x] = acc; __syncthreads();
    for (int s = 128; s > 0; s >>= 1) {
        if (threadIdx.x < s) red[threadIdx.x] += red[threadIdx.x + s];
        __syncthreads();
    }
    if (threadIdx.x == 0) csum[k] = red[0];
}

// =====================================================================
extern "C" void kernel_launch(void* const* d_in, const int* in_sizes, int n_in,
                              void* d_out, int out_size, void* d_ws, size_t ws_size,
                              hipStream_t stream) {
    (void)in_sizes; (void)n_in; (void)out_size; (void)ws_size;
    const float* x   = (const float*)d_in[0];   // (16384, 1024)
    const float* W   = (const float*)d_in[1];   // (1024, 1024)
    const int*   loc = (const int*)d_in[2];     // (1024, 2)
    const int*   itp = (const int*)d_in[3];     // scalar

    float* out  = (float*)d_out;
    float* loss = out;                           // d_out[0]
    float* newW = out + 1;                       // d_out[1..]

    char* wsb = (char*)d_ws; size_t off = 0;
    auto take = [&](size_t bytes) -> void* {
        off = (off + 255) & ~(size_t)255;
        void* p = wsb + off; off += bytes; return p;
    };
    __bf16* xbf     = (__bf16*)take((size_t)BATCH_N * IN_DIM * 2);
    __bf16* Wt      = (__bf16*)take((size_t)KNEUR * IN_DIM * 2);
    __bf16* Tb      = (__bf16*)take((size_t)KNEUR * KNEUR * 2);
    __bf16* St      = (__bf16*)take((size_t)IN_DIM * KNEUR * 2);
    float*  T       = (float*)take((size_t)KNEUR * KNEUR * 4);
    float*  S       = (float*)take((size_t)KNEUR * IN_DIM * 4);
    float*  x2      = (float*)take((size_t)BATCH_N * 4);
    float*  w2      = (float*)take((size_t)KNEUR * 4);
    float*  counts  = (float*)take((size_t)KNEUR * 4);
    float*  csum    = (float*)take((size_t)KNEUR * 4);
    float*  partval = (float*)take((size_t)16 * BATCH_N * 4);
    int*    partidx = (int*)take((size_t)16 * BATCH_N * 4);
    int*    bmu     = (int*)take((size_t)BATCH_N * 4);

    init_kernel<<<4101, 256, 0, stream>>>(S, counts, loss);
    prep_x_kernel<<<BATCH_N, 256, 0, stream>>>(x, xbf, x2);
    prep_w_kernel<<<KNEUR, 256, 0, stream>>>(W, Wt, w2);
    prep_T_kernel<<<4096, 256, 0, stream>>>(loc, itp, T, Tb);

    // GEMM1: distances + partial min/argmin. Tiles: N 8 blocks x M 128 blocks
    gemm_tile_kernel<0><<<dim3(8, 128), 256, 0, stream>>>(
        (const uint32_t*)xbf, (const uint32_t*)Wt, x2, w2, partval, partidx);

    combine_kernel<<<BATCH_N / 256, 256, 0, stream>>>(partval, partidx, bmu, loss);
    bucket_kernel<<<BATCH_N, 256, 0, stream>>>(x, bmu, S, counts);
    convS_kernel<<<4096, 256, 0, stream>>>(S, St);
    csum_kernel<<<KNEUR, 256, 0, stream>>>(T, counts, csum);

    // GEMM2: pos = S^T @ T, fused weight update epilogue
    gemm_tile_kernel<1><<<dim3(8, 8), 256, 0, stream>>>(
        (const uint32_t*)St, (const uint32_t*)Tb, W, csum, newW, (int*)nullptr);
}
